// Net_68453188764300
// MI455X (gfx1250) — matrix operand order repaired and checked by reading, not compiled, not run on hardware
//
#include <hip/hip_runtime.h>
#include <hip/hip_bf16.h>
#include <math.h>

#define N_NODES 20000
#define N_EDGES 40000
#define N_GRAPH 1000
#define F_IN    32
#define DIM     64
#define HIDE    128              // edge-MLP hidden width
#define KTOT    (DIM * HIDE)     // 8192 = GEMM K
#define KSTEPS  (KTOT / 32)      // 256 WMMA K-steps

typedef __attribute__((ext_vector_type(16))) __bf16 v16bf;
typedef __attribute__((ext_vector_type(8)))  float  v8f;

static_assert(sizeof(v16bf) == 32, "v16bf must be 32 bytes");

__device__ __forceinline__ float sigmoidf_(float x) { return 1.0f / (1.0f + __expf(-x)); }

__device__ __forceinline__ void atomicMaxF(float* addr, float val) {
    unsigned int* ua = (unsigned int*)addr;
    unsigned int old = *ua;
    while (__uint_as_float(old) < val) {
        unsigned int assumed = old;
        old = atomicCAS(ua, assumed, __float_as_uint(val));
        if (old == assumed) break;
    }
}

// ---------------------------------------------------------------- utilities
__global__ void fill_f32(float* p, float v, int n) {
    int i = blockIdx.x * blockDim.x + threadIdx.x;
    if (i < n) p[i] = v;
}

// ---------------------------------------------------------------- lin0 + relu
__global__ void lin0_k(const float* __restrict__ x, const float* __restrict__ W0,
                       const float* __restrict__ b0, float* __restrict__ out) {
    int i = blockIdx.x * blockDim.x + threadIdx.x;
    if (i >= N_NODES * DIM) return;
    int n = i / DIM, d = i % DIM;
    float acc = b0[d];
#pragma unroll
    for (int j = 0; j < F_IN; ++j) acc += x[n * F_IN + j] * W0[d * F_IN + j];
    out[i] = fmaxf(acc, 0.0f);
}

// ---------------------------------------------------------------- edge MLP layer 1 (bf16 output)
__global__ void edge_hidden_k(const float* __restrict__ ea, const float* __restrict__ We1,
                              const float* __restrict__ be1, __bf16* __restrict__ hid) {
    int i = blockIdx.x * blockDim.x + threadIdx.x;
    if (i >= N_EDGES * HIDE) return;
    int e = i / HIDE, k = i % HIDE;
    float acc = be1[k];
#pragma unroll
    for (int j = 0; j < 5; ++j) acc += ea[e * 5 + j] * We1[k * 5 + j];
    hid[i] = (__bf16)fmaxf(acc, 0.0f);
}

// ---------------------------------------------------------------- reorder We2 -> B-fragment-ordered W2rf (bf16)
// logical B: W2r[(i*128+k), o] = We2[(i*64+o)*128 + k],  K index = i*128+k
// fragment order: W2rf[ ((t*64 + o)*2 + h)*16 + j ] = B[ t*32 + h*16 + j ][ o ]
__global__ void reorder_w2_k(const float* __restrict__ We2, __bf16* __restrict__ W2rf) {
    int idx = blockIdx.x * blockDim.x + threadIdx.x;
    if (idx >= KTOT * DIM) return;
    int j = idx & 15;
    int h = (idx >> 4) & 1;
    int o = (idx >> 5) & 63;
    int t = idx >> 11;
    int i = t >> 2;
    int kb = (t & 3) * 32;
    W2rf[idx] = (__bf16)We2[(i * DIM + o) * HIDE + kb + h * 16 + j];
}

// ---------------------------------------------------------------- in-degree
__global__ void deg_k(const int* __restrict__ ei, float* __restrict__ deg) {
    int e = blockIdx.x * blockDim.x + threadIdx.x;
    if (e < N_EDGES) atomicAdd(&deg[ei[N_EDGES + e]], 1.0f);
}

// ---------------------------------------------------------------- fused message GEMM + bias + scatter-add
// per wave: 32 edges (2 M-tiles) x 64 outputs (4 N-tiles) -> 8 accumulators.
// msg = U[32,8192] @ W2r[8192,64], U built on the fly as x_src[e,i]*hid[e,k] (K=i*128+k).
// Two A fragments share each B fragment -> halves L2 traffic for the 1 MB B matrix.
__global__ void __launch_bounds__(64)
msg_gemm_k(const float* __restrict__ out, const __bf16* __restrict__ hid,
           const __bf16* __restrict__ W2rf, const float* __restrict__ be2,
           const int* __restrict__ ei, float* __restrict__ agg) {
    __shared__ __align__(16) float  xs[2][32][DIM];   // 16 KB
    __shared__ __align__(16) __bf16 hd[2][32][HIDE];  // 16 KB

    const int tid  = threadIdx.x;
    const int wave = tid >> 5;
    const int lane = tid & 31;
    const int e0   = blockIdx.x * 64 + wave * 32;

    // stage x_src (gather by src index) and hid tiles for this wave's 32 edges
    for (int idx = lane; idx < 32 * DIM; idx += 32) {
        int r = idx >> 6, c = idx & 63;
        int s = ei[e0 + r];
        xs[wave][r][c] = out[s * DIM + c];
    }
    for (int idx = lane; idx < 32 * HIDE; idx += 32) {
        int r = idx >> 7, c = idx & 127;
        hd[wave][r][c] = hid[(e0 + r) * HIDE + c];
    }
    __syncthreads();

    const int row  = lane & 15;   // A-matrix row within a 16-row tile / B column low bits
    const int half = lane >> 4;   // K-half selector (contiguous K order, same in A and B)
    const int k0   = half ? 16 : 0;

    v8f acc[2][4] = {};           // [m-tile][n-tile]
    const uint4* __restrict__ W4 = (const uint4*)W2rf;

    for (int t = 0; t < KSTEPS; ++t) {
        const int i  = t >> 2;
        const int kb = (t & 3) * 32;

        // two A fragments (edges row, row+16); lane<16 holds K {0..15}, lane>=16 {16..31}
        v16bf a[2];
#pragma unroll
        for (int mt = 0; mt < 2; ++mt) {
            const int rr = mt * 16 + row;
            const float xv = xs[wave][rr][i];
            union { uint4 u[2]; v16bf v; } ar;
            ar.u[0] = *(const uint4*)&hd[wave][rr][kb + k0];
            ar.u[1] = *(const uint4*)&hd[wave][rr][kb + k0 + 8];
#pragma unroll
            for (int j = 0; j < 16; ++j) a[mt][j] = (__bf16)(xv * (float)ar.v[j]);
        }

#pragma unroll
        for (int n = 0; n < 4; ++n) {
            const int col = n * 16 + row;
            union { uint4 u[2]; v16bf v; } b;
            const int base4 = ((t * 64 + col) * 2 + half) * 2;  // uint4 index
            b.u[0] = W4[base4];
            b.u[1] = W4[base4 + 1];
            acc[0][n] = __builtin_amdgcn_wmma_f32_16x16x32_bf16(
                false, a[0], false, b.v, (short)0, acc[0][n], false, false);
            acc[1][n] = __builtin_amdgcn_wmma_f32_16x16x32_bf16(
                false, a[1], false, b.v, (short)0, acc[1][n], false, false);
        }
    }

    // epilogue per m-tile: bias  Σ_i x_src[m,i]*be2[i*64+o]  then scatter-add into agg[dst]
#pragma unroll
    for (int mt = 0; mt < 2; ++mt) {
        float bias[4][8];
#pragma unroll
        for (int n = 0; n < 4; ++n)
#pragma unroll
            for (int r = 0; r < 8; ++r) bias[n][r] = 0.0f;

        for (int i2 = 0; i2 < DIM; ++i2) {
            float w[4];
#pragma unroll
            for (int n = 0; n < 4; ++n) w[n] = be2[i2 * DIM + n * 16 + row];
#pragma unroll
            for (int r = 0; r < 8; ++r) {
                const float xr = xs[wave][mt * 16 + half * 8 + r][i2];
#pragma unroll
                for (int n = 0; n < 4; ++n) bias[n][r] += xr * w[n];
            }
        }

#pragma unroll
        for (int r = 0; r < 8; ++r) {
            const int m = mt * 16 + half * 8 + r;   // C/D layout: lanes>=16 hold M=r+8
            const int d = ei[N_EDGES + e0 + m];     // dst node
#pragma unroll
            for (int n = 0; n < 4; ++n) {
                atomicAdd(&agg[d * DIM + n * 16 + row], acc[mt][n][r] + bias[n][r]);
            }
        }
    }
}

// ---------------------------------------------------------------- GRU cell (one block = one node)
__global__ void gru_k(const float* __restrict__ agg, const float* __restrict__ deg,
                      const float* __restrict__ b_conv,
                      const float* __restrict__ W_ih, const float* __restrict__ W_hh,
                      const float* __restrict__ b_ih, const float* __restrict__ b_hh,
                      float* __restrict__ h) {
    __shared__ float m[DIM], hh[DIM];
    const int n = blockIdx.x, d = threadIdx.x;
    const float dg = fmaxf(deg[n], 1.0f);
    m[d]  = fmaxf(agg[n * DIM + d] / dg + b_conv[d], 0.0f);
    hh[d] = h[n * DIM + d];
    __syncthreads();
    float ir = b_ih[d], iz = b_ih[DIM + d], in_ = b_ih[2 * DIM + d];
    float hr = b_hh[d], hz = b_hh[DIM + d], hn  = b_hh[2 * DIM + d];
    for (int j = 0; j < DIM; ++j) {
        const float mj = m[j], hj = hh[j];
        ir  += mj * W_ih[d * DIM + j];
        iz  += mj * W_ih[(DIM + d) * DIM + j];
        in_ += mj * W_ih[(2 * DIM + d) * DIM + j];
        hr  += hj * W_hh[d * DIM + j];
        hz  += hj * W_hh[(DIM + d) * DIM + j];
        hn  += hj * W_hh[(2 * DIM + d) * DIM + j];
    }
    const float r  = sigmoidf_(ir + hr);
    const float z  = sigmoidf_(iz + hz);
    const float ng = tanhf(in_ + r * hn);
    h[n * DIM + d] = (1.0f - z) * ng + z * hh[d];
}

// ---------------------------------------------------------------- Set2Set LSTM (one block = one graph)
__global__ void lstm_k(const float* __restrict__ q_star,
                       const float* __restrict__ W_ihl, const float* __restrict__ W_hhl,
                       const float* __restrict__ b_ihl, const float* __restrict__ b_hhl,
                       float* __restrict__ hl, float* __restrict__ cl) {
    __shared__ float qs[2 * DIM], hs[DIM];
    const int b = blockIdx.x, d = threadIdx.x;
    qs[d]       = q_star[b * 2 * DIM + d];
    qs[DIM + d] = q_star[b * 2 * DIM + DIM + d];
    hs[d]       = hl[b * DIM + d];
    __syncthreads();
    float g[4];
#pragma unroll
    for (int gi = 0; gi < 4; ++gi) {
        const int rowi = gi * DIM + d;
        float acc = b_ihl[rowi] + b_hhl[rowi];
        for (int j = 0; j < 2 * DIM; ++j) acc += qs[j] * W_ihl[rowi * 2 * DIM + j];
        for (int j = 0; j < DIM; ++j)     acc += hs[j] * W_hhl[rowi * DIM + j];
        g[gi] = acc;
    }
    const float c_new = sigmoidf_(g[1]) * cl[b * DIM + d] + sigmoidf_(g[0]) * tanhf(g[2]);
    const float h_new = sigmoidf_(g[3]) * tanhf(c_new);
    cl[b * DIM + d] = c_new;
    hl[b * DIM + d] = h_new;
}

__global__ void stats_init_k(float* amax, float* asum, float* rvec) {
    int i = blockIdx.x * blockDim.x + threadIdx.x;
    if (i < N_GRAPH * DIM) rvec[i] = 0.0f;
    if (i < N_GRAPH) { amax[i] = -3.0e38f; asum[i] = 0.0f; }
}

__global__ void attn_e_k(const float* __restrict__ out, const float* __restrict__ hl,
                         const int* __restrict__ batch, float* __restrict__ ebuf,
                         float* __restrict__ amax) {
    int n = blockIdx.x * blockDim.x + threadIdx.x;
    if (n >= N_NODES) return;
    const int b = batch[n];
    float acc = 0.0f;
    for (int j = 0; j < DIM; ++j) acc += out[n * DIM + j] * hl[b * DIM + j];
    ebuf[n] = acc;
    atomicMaxF(&amax[b], acc);
}

__global__ void attn_exp_k(const int* __restrict__ batch, const float* __restrict__ amax,
                           float* __restrict__ ebuf, float* __restrict__ asum) {
    int n = blockIdx.x * blockDim.x + threadIdx.x;
    if (n >= N_NODES) return;
    const int b = batch[n];
    const float a = __expf(ebuf[n] - amax[b]);
    ebuf[n] = a;
    atomicAdd(&asum[b], a);
}

__global__ void attn_r_k(const float* __restrict__ out, const int* __restrict__ batch,
                         const float* __restrict__ ebuf, const float* __restrict__ asum,
                         float* __restrict__ rvec) {
    int idx = blockIdx.x * blockDim.x + threadIdx.x;
    if (idx >= N_NODES * DIM) return;
    const int n = idx >> 6, d = idx & 63;
    const int b = batch[n];
    const float w = ebuf[n] / asum[b];
    atomicAdd(&rvec[b * DIM + d], w * out[idx]);
}

__global__ void qstar_update_k(const float* __restrict__ hl, const float* __restrict__ rvec,
                               float* __restrict__ q_star) {
    int idx = blockIdx.x * blockDim.x + threadIdx.x;
    if (idx >= N_GRAPH * 2 * DIM) return;
    const int b = idx >> 7, d = idx & 127;
    q_star[idx] = (d < DIM) ? hl[b * DIM + d] : rvec[b * DIM + d - DIM];
}

// ---------------------------------------------------------------- final MLP (one block = one graph)
__global__ void final_k(const float* __restrict__ q_star, const float* __restrict__ W1,
                        const float* __restrict__ b1, const float* __restrict__ W2,
                        const float* __restrict__ b2, float* __restrict__ pred) {
    __shared__ float y[DIM];
    const int b = blockIdx.x, d = threadIdx.x;
    float acc = b1[d];
    for (int j = 0; j < 2 * DIM; ++j) acc += q_star[b * 2 * DIM + j] * W1[d * 2 * DIM + j];
    y[d] = fmaxf(acc, 0.0f);
    __syncthreads();
    if (d == 0) {
        float s = b2[0];
        for (int j = 0; j < DIM; ++j) s += y[j] * W2[j];
        pred[b] = s;
    }
}

// ---------------------------------------------------------------- host driver
extern "C" void kernel_launch(void* const* d_in, const int* in_sizes, int n_in,
                              void* d_out, int out_size, void* d_ws, size_t ws_size,
                              hipStream_t stream) {
    const float* x       = (const float*)d_in[0];
    const float* ea      = (const float*)d_in[1];
    const int*   ei      = (const int*)  d_in[2];
    const int*   batch   = (const int*)  d_in[3];
    const float* W0      = (const float*)d_in[4];
    const float* b0      = (const float*)d_in[5];
    const float* We1     = (const float*)d_in[6];
    const float* be1     = (const float*)d_in[7];
    const float* We2     = (const float*)d_in[8];
    const float* be2     = (const float*)d_in[9];
    const float* b_conv  = (const float*)d_in[10];
    const float* W_ih    = (const float*)d_in[11];
    const float* W_hh    = (const float*)d_in[12];
    const float* b_ih    = (const float*)d_in[13];
    const float* b_hh    = (const float*)d_in[14];
    const float* W_ihl   = (const float*)d_in[15];
    const float* W_hhl   = (const float*)d_in[16];
    const float* b_ihl   = (const float*)d_in[17];
    const float* b_hhl   = (const float*)d_in[18];
    const float* W1      = (const float*)d_in[19];
    const float* b1      = (const float*)d_in[20];
    const float* W2      = (const float*)d_in[21];
    const float* b2      = (const float*)d_in[22];
    float* pred = (float*)d_out;

    // workspace layout
    char* base = (char*)d_ws;
    size_t off = 0;
    auto alloc = [&](size_t bytes) -> void* {
        void* p = base + off;
        off = (off + bytes + 255) & ~(size_t)255;
        return p;
    };
    float*  out_h = (float*) alloc((size_t)N_NODES * DIM * 4);
    float*  agg   = (float*) alloc((size_t)N_NODES * DIM * 4);
    float*  deg   = (float*) alloc((size_t)N_NODES * 4);
    __bf16* hid   = (__bf16*)alloc((size_t)N_EDGES * HIDE * 2);
    __bf16* W2rf  = (__bf16*)alloc((size_t)KTOT * DIM * 2);
    float*  ebuf  = (float*) alloc((size_t)N_NODES * 4);
    float*  amax  = (float*) alloc((size_t)N_GRAPH * 4);
    float*  asum  = (float*) alloc((size_t)N_GRAPH * 4);
    float*  hl    = (float*) alloc((size_t)N_GRAPH * DIM * 4);
    float*  cl    = (float*) alloc((size_t)N_GRAPH * DIM * 4);
    float*  rvec  = (float*) alloc((size_t)N_GRAPH * DIM * 4);
    float*  qstar = (float*) alloc((size_t)N_GRAPH * 2 * DIM * 4);
    (void)ws_size; (void)n_in; (void)in_sizes; (void)out_size;

    const int TB = 256;
    auto grid = [](int n, int tb) { return (n + tb - 1) / tb; };

    // preprocessing (every call — deterministic)
    lin0_k<<<grid(N_NODES * DIM, TB), TB, 0, stream>>>(x, W0, b0, out_h);
    edge_hidden_k<<<grid(N_EDGES * HIDE, TB), TB, 0, stream>>>(ea, We1, be1, hid);
    reorder_w2_k<<<grid(KTOT * DIM, TB), TB, 0, stream>>>(We2, W2rf);
    fill_f32<<<grid(N_NODES, TB), TB, 0, stream>>>(deg, 0.0f, N_NODES);
    deg_k<<<grid(N_EDGES, TB), TB, 0, stream>>>(ei, deg);

    // 3 message-passing + GRU iterations
    for (int it = 0; it < 3; ++it) {
        fill_f32<<<grid(N_NODES * DIM, TB), TB, 0, stream>>>(agg, 0.0f, N_NODES * DIM);
        msg_gemm_k<<<N_EDGES / 64, 64, 0, stream>>>(out_h, hid, W2rf, be2, ei, agg);
        gru_k<<<N_NODES, DIM, 0, stream>>>(agg, deg, b_conv, W_ih, W_hh, b_ih, b_hh, out_h);
    }

    // Set2Set pooling
    fill_f32<<<grid(N_GRAPH * 2 * DIM, TB), TB, 0, stream>>>(qstar, 0.0f, N_GRAPH * 2 * DIM);
    fill_f32<<<grid(N_GRAPH * DIM, TB), TB, 0, stream>>>(hl, 0.0f, N_GRAPH * DIM);
    fill_f32<<<grid(N_GRAPH * DIM, TB), TB, 0, stream>>>(cl, 0.0f, N_GRAPH * DIM);
    for (int s = 0; s < 3; ++s) {
        lstm_k<<<N_GRAPH, DIM, 0, stream>>>(qstar, W_ihl, W_hhl, b_ihl, b_hhl, hl, cl);
        stats_init_k<<<grid(N_GRAPH * DIM, TB), TB, 0, stream>>>(amax, asum, rvec);
        attn_e_k<<<grid(N_NODES, TB), TB, 0, stream>>>(out_h, hl, batch, ebuf, amax);
        attn_exp_k<<<grid(N_NODES, TB), TB, 0, stream>>>(batch, amax, ebuf, asum);
        attn_r_k<<<grid(N_NODES * DIM, TB), TB, 0, stream>>>(out_h, batch, ebuf, asum, rvec);
        qstar_update_k<<<grid(N_GRAPH * 2 * DIM, TB), TB, 0, stream>>>(hl, rvec, qstar);
    }

    final_k<<<N_GRAPH, DIM, 0, stream>>>(qstar, W1, b1, W2, b2, pred);
}
